// SpatialMamba_15092515078212
// MI455X (gfx1250) — matrix operand order repaired and checked by reading, not compiled, not run on hardware
//
#include <hip/hip_runtime.h>
#include <hip/hip_bf16.h>

// ---------------- CDNA5 WMMA types ----------------
typedef __attribute__((ext_vector_type(16))) __bf16 v16bf;
typedef __attribute__((ext_vector_type(8)))  float  v8f;

union BfVec { v16bf v; unsigned short u[16]; };

__device__ __forceinline__ unsigned short f2bf(float f) {
  union { float f; unsigned u; } c; c.f = f;
  unsigned r = c.u + 0x7FFFu + ((c.u >> 16) & 1u);   // round-to-nearest-even
  return (unsigned short)(r >> 16);
}
__device__ __forceinline__ float sigmoidf(float x) { return 1.f / (1.f + __expf(-x)); }

#define L_TOT  131072
#define HW     65536
#define CIN    32
#define DIN    64
#define TL1    64
#define NCHUNK 512
#define CL     256
#define SUBL   64

// ---------------- K0: zero GN accumulators ----------------
__global__ void k_zero(float* p, int n) {
  int i = blockIdx.x * blockDim.x + threadIdx.x;
  if (i < n) p[i] = 0.f;
}

// ---------------- K1: transpose-in + WMMA GEMM1 + conv + SiLU + xproj + softplus ----------------
__global__ __launch_bounds__(256) void k_front(
    const float* __restrict__ x, const float* __restrict__ W_in,
    const float* __restrict__ conv_w, const float* __restrict__ conv_b,
    const float* __restrict__ W_xproj, const float* __restrict__ W_dt,
    const float* __restrict__ b_dt,
    float* __restrict__ zg, float* __restrict__ ucg,
    float* __restrict__ deltag, float* __restrict__ bcg)
{
  __shared__ unsigned short wt[128][CIN];      // W_in^T bf16: wt[n][c]
  __shared__ unsigned short xbf[TL1][CIN];     // xf bf16 tile
  __shared__ float u_lds[TL1 + 3][DIN];        // conv input incl. 3-row halo
  __shared__ float uc_lds[TL1][DIN];
  __shared__ float hx[3][CIN];
  __shared__ float dt_lds[TL1][2];

  const int t  = threadIdx.x;
  const int l0 = blockIdx.x * TL1;

  // W_in -> LDS (transposed, bf16)
  for (int p = t; p < CIN * 128; p += 256) {
    int c = p >> 7, n = p & 127;
    wt[n][c] = f2bf(W_in[p]);
  }
  // xf tile (NCHW -> (L,C) transpose read, coalesced per channel)
  {
    int b = l0 >> 16;                 // HW % TL1 == 0: tile stays in one batch
    int rem0 = l0 & (HW - 1);
    for (int p = t; p < TL1 * CIN; p += 256) {
      int c = p >> 6, r = p & 63;
      xbf[r][c] = f2bf(x[(size_t)b * CIN * HW + (size_t)c * HW + rem0 + r]);
    }
  }
  // halo xf rows (fp32)
  if (t < 3 * CIN) {
    int k = t / CIN, c = t % CIN;
    int l = l0 - 3 + k;
    float v = 0.f;
    if (l >= 0) v = x[(size_t)(l >> 16) * CIN * HW + (size_t)c * HW + (l & (HW - 1))];
    hx[k][c] = v;
  }
  __syncthreads();

  // WMMA: xz = xf @ W_in   (64x32 @ 32x128), K=32 in one v_wmma_f32_16x16x32_bf16
  {
    const int lane = t & 31;
    // wave index as a wave-uniform scalar -> s_cmp/s_cbranch instead of exec juggling
    const int wave = __builtin_amdgcn_readfirstlane(t >> 5);
    const int m16 = lane & 15, g = lane >> 4;
    const int n0 = wave * 16;
    BfVec bv;
#pragma unroll
    for (int j = 0; j < 16; ++j) bv.u[j] = wt[n0 + m16][(j >> 3) * 16 + g * 8 + (j & 7)];
    if (wave < 4) {
      // u half: results go to LDS for the conv
      const int ch = n0 + m16;
#pragma unroll
      for (int Mt = 0; Mt < TL1 / 16; ++Mt) {
        BfVec av;
#pragma unroll
        for (int j = 0; j < 8; ++j) {
          av.u[j]     = xbf[Mt * 16 + m16][g * 8 + j];
          av.u[8 + j] = xbf[Mt * 16 + m16][16 + g * 8 + j];
        }
        v8f acc = {};
        acc = __builtin_amdgcn_wmma_f32_16x16x32_bf16(false, av.v, false, bv.v,
                                                      (short)0, acc, false, false);
#pragma unroll
        for (int rr = 0; rr < 8; ++rr)
          u_lds[3 + Mt * 16 + g * 8 + rr][ch] = acc[rr];
      }
    } else {
      // z half: straight to global
      const int zc = n0 + m16 - DIN;
#pragma unroll
      for (int Mt = 0; Mt < TL1 / 16; ++Mt) {
        BfVec av;
#pragma unroll
        for (int j = 0; j < 8; ++j) {
          av.u[j]     = xbf[Mt * 16 + m16][g * 8 + j];
          av.u[8 + j] = xbf[Mt * 16 + m16][16 + g * 8 + j];
        }
        v8f acc = {};
        acc = __builtin_amdgcn_wmma_f32_16x16x32_bf16(false, av.v, false, bv.v,
                                                      (short)0, acc, false, false);
#pragma unroll
        for (int rr = 0; rr < 8; ++rr)
          zg[(size_t)(l0 + Mt * 16 + g * 8 + rr) * DIN + zc] = acc[rr];
      }
    }
  }
  // halo u rows (exact fp32 dot)
  if (t < 3 * DIN) {
    int k = t / DIN, ch = t % DIN;
    int l = l0 - 3 + k;
    float s = 0.f;
    if (l >= 0)
      for (int c = 0; c < CIN; ++c) s += hx[k][c] * W_in[c * 128 + ch];
    u_lds[k][ch] = s;
  }
  __syncthreads();

  // causal depthwise conv (k=4) + SiLU
  for (int p = t; p < TL1 * DIN; p += 256) {
    int r = p >> 6, ch = p & 63;
    float v = conv_b[ch];
#pragma unroll
    for (int k = 0; k < 4; ++k) v += u_lds[r + k][ch] * conv_w[ch * 4 + k];
    v = v * sigmoidf(v);
    uc_lds[r][ch] = v;
    ucg[(size_t)(l0 + r) * DIN + ch] = v;
  }
  __syncthreads();

  // dbc = uc @ W_xproj  (K=64, N=34): dt(2) -> LDS, B(16)/C(16) -> global
  {
    int r = t >> 2, sub = t & 3;
    int o0 = sub * 9;
    int o1 = (o0 + 9 < 34) ? o0 + 9 : 34;
    for (int o = o0; o < o1; ++o) {
      float s = 0.f;
      for (int c = 0; c < DIN; ++c) s += uc_lds[r][c] * W_xproj[c * 34 + o];
      if (o < 2) dt_lds[r][o] = s;
      else       bcg[(size_t)(l0 + r) * 32 + (o - 2)] = s;
    }
  }
  __syncthreads();

  // delta = softplus(dt @ W_dt + b_dt)
  for (int p = t; p < TL1 * DIN; p += 256) {
    int r = p >> 6, ch = p & 63;
    float v = dt_lds[r][0] * W_dt[ch] + dt_lds[r][1] * W_dt[DIN + ch] + b_dt[ch];
    float sp = (v > 20.f) ? v : log1pf(__expf(v));
    deltag[(size_t)(l0 + r) * DIN + ch] = sp;
  }
}

// ---------------- K2: scan phase 1 (per-chunk state from h0=0, plus sum(delta)) ----------------
__global__ __launch_bounds__(256) void k_scan1(
    const float* __restrict__ deltag, const float* __restrict__ ucg,
    const float* __restrict__ bcg, const float* __restrict__ A_log,
    float* __restrict__ hend, float* __restrict__ sumd)
{
  __shared__ float sd[SUBL][DIN];
  __shared__ float su[SUBL][DIN];
  __shared__ float sB[SUBL][16];

  const int t = threadIdx.x;
  const int chunk = blockIdx.x;
  const int ch = t >> 2, sg = t & 3;
  float Av[4], h[4];
#pragma unroll
  for (int i = 0; i < 4; ++i) {
    Av[i] = -__expf(A_log[ch * 16 + sg * 4 + i]);
    h[i] = 0.f;
  }
  float sdsum = 0.f;
  for (int sub = 0; sub < CL / SUBL; ++sub) {
    int lb = chunk * CL + sub * SUBL;
    for (int p = t; p < SUBL * DIN; p += 256) {
      int rr = p >> 6, cc = p & 63;
      sd[rr][cc] = deltag[(size_t)(lb + rr) * DIN + cc];
      su[rr][cc] = ucg[(size_t)(lb + rr) * DIN + cc];
    }
    for (int p = t; p < SUBL * 16; p += 256) {
      int rr = p >> 4, ss = p & 15;
      sB[rr][ss] = bcg[(size_t)(lb + rr) * 32 + ss];
    }
    __syncthreads();
    for (int rr = 0; rr < SUBL; ++rr) {
      float d = sd[rr][ch];
      float du = d * su[rr][ch];
      sdsum += d;
#pragma unroll
      for (int i = 0; i < 4; ++i)
        h[i] = __expf(d * Av[i]) * h[i] + du * sB[rr][sg * 4 + i];
    }
    __syncthreads();
  }
#pragma unroll
  for (int i = 0; i < 4; ++i)
    hend[(size_t)chunk * 1024 + ch * 16 + sg * 4 + i] = h[i];
  if (sg == 0) sumd[chunk * 64 + ch] = sdsum;
}

// ---------------- K3: scan phase 2 (sequential prefix over 512 chunks) ----------------
__global__ __launch_bounds__(1024) void k_scan2(
    const float* __restrict__ hend, const float* __restrict__ sumd,
    const float* __restrict__ A_log, float* __restrict__ hin)
{
  const int t = threadIdx.x;           // t = ch*16 + st
  const int ch = t >> 4;
  const float Acs = -__expf(A_log[t]);
  float h = 0.f;
  for (int k = 0; k < NCHUNK; ++k) {
    hin[(size_t)k * 1024 + t] = h;
    h = __expf(Acs * sumd[k * 64 + ch]) * h + hend[(size_t)k * 1024 + t];
  }
}

// ---------------- K4: scan phase 3 (rescan with true h_in; fuse y, D*uc, z-gate; in-place) ----------------
__global__ __launch_bounds__(256) void k_scan3(
    const float* __restrict__ deltag, float* __restrict__ ucg,   // y overwrites uc
    const float* __restrict__ bcg, const float* __restrict__ zg,
    const float* __restrict__ A_log, const float* __restrict__ Dp,
    const float* __restrict__ hin)
{
  __shared__ float sd[SUBL][DIN];
  __shared__ float su[SUBL][DIN];
  __shared__ float sz[SUBL][DIN];
  __shared__ float sB[SUBL][16];
  __shared__ float sC[SUBL][16];

  const int t = threadIdx.x;
  const int chunk = blockIdx.x;
  const int ch = t >> 2, sg = t & 3;
  float Av[4], h[4];
#pragma unroll
  for (int i = 0; i < 4; ++i) {
    Av[i] = -__expf(A_log[ch * 16 + sg * 4 + i]);
    h[i] = hin[(size_t)chunk * 1024 + ch * 16 + sg * 4 + i];
  }
  const float dpc = Dp[ch];
  for (int sub = 0; sub < CL / SUBL; ++sub) {
    int lb = chunk * CL + sub * SUBL;
    for (int p = t; p < SUBL * DIN; p += 256) {
      int rr = p >> 6, cc = p & 63;
      sd[rr][cc] = deltag[(size_t)(lb + rr) * DIN + cc];
      su[rr][cc] = ucg[(size_t)(lb + rr) * DIN + cc];
      sz[rr][cc] = zg[(size_t)(lb + rr) * DIN + cc];
    }
    for (int p = t; p < SUBL * 16; p += 256) {
      int rr = p >> 4, ss = p & 15;
      sB[rr][ss] = bcg[(size_t)(lb + rr) * 32 + ss];
      sC[rr][ss] = bcg[(size_t)(lb + rr) * 32 + 16 + ss];
    }
    __syncthreads();
    for (int rr = 0; rr < SUBL; ++rr) {
      float d = sd[rr][ch];
      float u = su[rr][ch];
      float du = d * u;
      float yp = 0.f;
#pragma unroll
      for (int i = 0; i < 4; ++i) {
        h[i] = __expf(d * Av[i]) * h[i] + du * sB[rr][sg * 4 + i];
        yp += h[i] * sC[rr][sg * 4 + i];
      }
      yp += __shfl_xor(yp, 1);     // wave32: 4 lanes per channel
      yp += __shfl_xor(yp, 2);
      if (sg == 0) {
        float y = yp + dpc * u;
        float zv = sz[rr][ch];
        y *= zv * sigmoidf(zv);
        ucg[(size_t)(lb + rr) * DIN + ch] = y;
      }
    }
    __syncthreads();
  }
}

// ---------------- K5: WMMA out-GEMM (Lx64 @ 64x32) + GroupNorm partial sums ----------------
__global__ __launch_bounds__(256) void k_outgemm(
    const float* __restrict__ yg, const float* __restrict__ W_out,
    float* __restrict__ outp, float* __restrict__ gsum, float* __restrict__ gsumsq)
{
  __shared__ unsigned short ybf[128][DIN];
  __shared__ unsigned short wo[32][DIN];      // W_out^T bf16: wo[n][c]
  const int t = threadIdx.x;
  const int l0 = blockIdx.x * 128;
  for (int p = t; p < DIN * 32; p += 256) {
    int c = p >> 5, n = p & 31;
    wo[n][c] = f2bf(W_out[p]);
  }
  for (int p = t; p < 128 * DIN; p += 256) {
    int r = p >> 6, c = p & 63;
    ybf[r][c] = f2bf(yg[(size_t)(l0 + r) * DIN + c]);
  }
  __syncthreads();
  const int lane = t & 31;
  const int Mt = __builtin_amdgcn_readfirstlane(t >> 5);   // wave-uniform M tile
  const int m16 = lane & 15, g = lane >> 4;
  const int b = l0 >> 16;
#pragma unroll
  for (int Nt = 0; Nt < 2; ++Nt) {
    v8f acc = {};
#pragma unroll
    for (int ka = 0; ka < 2; ++ka) {         // K = 64 in two x32 steps
      BfVec av, bv;
#pragma unroll
      for (int j = 0; j < 8; ++j) {
        av.u[j]     = ybf[Mt * 16 + m16][ka * 32 + g * 8 + j];
        av.u[8 + j] = ybf[Mt * 16 + m16][ka * 32 + 16 + g * 8 + j];
      }
#pragma unroll
      for (int j = 0; j < 16; ++j) bv.u[j] = wo[Nt * 16 + m16][ka * 32 + (j >> 3) * 16 + g * 8 + (j & 7)];
      acc = __builtin_amdgcn_wmma_f32_16x16x32_bf16(false, av.v, false, bv.v,
                                                    (short)0, acc, false, false);
    }
    int col = Nt * 16 + m16;
    float s1 = 0.f, s2 = 0.f;
#pragma unroll
    for (int rr = 0; rr < 8; ++rr) {
      int r = Mt * 16 + g * 8 + rr;
      float v = acc[rr];
      outp[(size_t)(l0 + r) * 32 + col] = v;
      s1 += v; s2 += v * v;
    }
    // reduce over 8 lanes of same group and both row halves (g): masks 1,2,4,16
    s1 += __shfl_xor(s1, 1);  s2 += __shfl_xor(s2, 1);
    s1 += __shfl_xor(s1, 2);  s2 += __shfl_xor(s2, 2);
    s1 += __shfl_xor(s1, 4);  s2 += __shfl_xor(s2, 4);
    s1 += __shfl_xor(s1, 16); s2 += __shfl_xor(s2, 16);
    if (lane == 0 || lane == 8) {
      int grp = Nt * 2 + (m16 >> 3);
      atomicAdd(&gsum[b * 4 + grp], s1);
      atomicAdd(&gsumsq[b * 4 + grp], s2);
    }
  }
}

// ---------------- K6: GroupNorm + SiLU + residual, (L,C) -> NCHW via LDS transpose ----------------
__global__ __launch_bounds__(256) void k_final(
    const float* __restrict__ outp, const float* __restrict__ x,
    const float* __restrict__ gamma, const float* __restrict__ beta,
    const float* __restrict__ gsum, const float* __restrict__ gsumsq,
    float* __restrict__ out)
{
  __shared__ float tile[64][33];
  const int t = threadIdx.x;
  const int l0 = blockIdx.x * 64;
  const int b = l0 >> 16;
  const int rem0 = l0 & (HW - 1);
  const float inv_cnt = 1.f / (8.f * (float)HW);
  for (int p = t; p < 64 * 32; p += 256) {
    int rr = p >> 5, c = p & 31;
    int grp = c >> 3;
    float mu = gsum[b * 4 + grp] * inv_cnt;
    float var = gsumsq[b * 4 + grp] * inv_cnt - mu * mu;
    float rs = rsqrtf(var + 1e-5f);
    float v = outp[(size_t)(l0 + rr) * 32 + c];
    v = (v - mu) * rs * gamma[c] + beta[c];
    v = v * sigmoidf(v);
    tile[rr][c] = v;
  }
  __syncthreads();
  for (int p = t; p < 64 * 32; p += 256) {
    int c = p >> 6, rr = p & 63;
    size_t ix = (size_t)b * CIN * HW + (size_t)c * HW + rem0 + rr;
    out[ix] = tile[rr][c] + x[ix];
  }
}

// ---------------- launch ----------------
extern "C" void kernel_launch(void* const* d_in, const int* in_sizes, int n_in,
                              void* d_out, int out_size, void* d_ws, size_t ws_size,
                              hipStream_t stream) {
  (void)in_sizes; (void)n_in; (void)out_size; (void)ws_size;
  const float* x      = (const float*)d_in[0];
  const float* W_in   = (const float*)d_in[1];
  const float* conv_w = (const float*)d_in[2];
  const float* conv_b = (const float*)d_in[3];
  const float* W_xproj= (const float*)d_in[4];
  const float* W_dt   = (const float*)d_in[5];
  const float* b_dt   = (const float*)d_in[6];
  const float* A_log  = (const float*)d_in[7];
  const float* Dp     = (const float*)d_in[8];
  const float* W_out  = (const float*)d_in[9];
  const float* gamma  = (const float*)d_in[10];
  const float* beta   = (const float*)d_in[11];
  float* out = (float*)d_out;

  const size_t L = L_TOT;
  float* ws    = (float*)d_ws;
  float* zg    = ws;                       // L*64
  float* ucg   = zg + L * 64;              // L*64 (y in-place after scan3)
  float* dg    = ucg + L * 64;             // L*64
  float* bcg   = dg + L * 64;              // L*32 (Bm|Cm; reused as out_pre by K5/K6)
  float* hend  = bcg + L * 32;             // 512*1024
  float* sumd  = hend + (size_t)NCHUNK * 1024;  // 512*64
  float* hin   = sumd + (size_t)NCHUNK * 64;    // 512*1024
  float* gsum  = hin + (size_t)NCHUNK * 1024;   // 16
  float* gsumsq= gsum + 16;                     // 16

  k_zero   <<<1, 32, 0, stream>>>(gsum, 32);
  k_front  <<<L / TL1, 256, 0, stream>>>(x, W_in, conv_w, conv_b, W_xproj, W_dt, b_dt,
                                         zg, ucg, dg, bcg);
  k_scan1  <<<NCHUNK, 256, 0, stream>>>(dg, ucg, bcg, A_log, hend, sumd);
  k_scan2  <<<1, 1024, 0, stream>>>(hend, sumd, A_log, hin);
  k_scan3  <<<NCHUNK, 256, 0, stream>>>(dg, ucg, bcg, zg, A_log, Dp, hin);
  k_outgemm<<<L / 128, 256, 0, stream>>>(ucg, W_out, bcg, gsum, gsumsq);
  k_final  <<<L / 64, 256, 0, stream>>>(bcg, x, gamma, beta, gsum, gsumsq, out);
}
